// VectorQuantizeLLM_21303037788419
// MI455X (gfx1250) — compile-verified, matrix operand order checked by reading, not run on hardware
//
#include <hip/hip_runtime.h>
#include <hip/hip_bf16.h>

// Problem constants (match reference)
#define BB  4
#define DIM 1024
#define TT  512
#define VV  32000
#define CDI 4096
#define EPSN 1e-12f

// ---- plain ext-vector types (trivially constructible; safe in unions) ----
typedef __attribute__((ext_vector_type(16))) __bf16       v16bf;
typedef __attribute__((ext_vector_type(8)))  float        v8f;
typedef __attribute__((ext_vector_type(4)))  float        f32x4;
typedef __attribute__((ext_vector_type(4)))  unsigned int u32x4;
typedef __attribute__((ext_vector_type(2)))  unsigned int u32x2;

union FragU { v16bf v; u32x4 q[2]; unsigned short h[16]; };

__device__ __forceinline__ unsigned short f2bf(float f) {
  unsigned int u = __float_as_uint(f);
  u += 0x7FFFu + ((u >> 16) & 1u);        // round-to-nearest-even
  return (unsigned short)(u >> 16);
}

// Ordered-uint mapping for packed float atomicMax
__device__ __forceinline__ unsigned int ford(float f) {
  unsigned int u = __float_as_uint(f);
  return (u & 0x80000000u) ? ~u : (u | 0x80000000u);
}

// NT fragment loader: row-major [rows][ld] bf16, ISA 16x32 bf16 A/B layout
// (lane&15 selects the row/col, lane>>4 selects which 8-wide K half).
__device__ __forceinline__ v16bf ldfrag_bf16(const unsigned short* base, int row0,
                                             int ld, int k0, int lane) {
  int r = lane & 15, hi = (lane >> 4) & 1;
  const unsigned short* p = base + (size_t)(row0 + r) * ld + (k0 + hi * 8);
  FragU u;
  u.q[0] = *(const u32x4*)p;          // K = k0+hi*8   .. +7
  u.q[1] = *(const u32x4*)(p + 16);   // K = k0+16+hi*8.. +7
  return u.v;
}

// Same fragment but fp32 source: convert in registers (VALU co-executes with
// the XDL WMMA pipe), saving a 262MB bf16 staging pass for the codebook.
__device__ __forceinline__ v16bf ldfrag_f32(const float* base, int row0,
                                            int ld, int k0, int lane) {
  int r = lane & 15, hi = (lane >> 4) & 1;
  const float* p = base + (size_t)(row0 + r) * ld + (k0 + hi * 8);
  f32x4 a = *(const f32x4*)p;
  f32x4 b = *(const f32x4*)(p + 4);
  f32x4 c = *(const f32x4*)(p + 16);
  f32x4 d = *(const f32x4*)(p + 20);
  FragU u;
  u.h[0] = f2bf(a[0]); u.h[1] = f2bf(a[1]); u.h[2]  = f2bf(a[2]); u.h[3]  = f2bf(a[3]);
  u.h[4] = f2bf(b[0]); u.h[5] = f2bf(b[1]); u.h[6]  = f2bf(b[2]); u.h[7]  = f2bf(b[3]);
  u.h[8] = f2bf(c[0]); u.h[9] = f2bf(c[1]); u.h[10] = f2bf(c[2]); u.h[11] = f2bf(c[3]);
  u.h[12]= f2bf(d[0]); u.h[13]= f2bf(d[1]); u.h[14] = f2bf(d[2]); u.h[15] = f2bf(d[3]);
  return u.v;
}

#define WMMA_BF16(A, B, C) __builtin_amdgcn_wmma_f32_16x16x32_bf16( \
    false, (A), false, (B), (short)0, (C), false, false)

// ---------- small prep kernels ----------

// fp32 -> bf16 elementwise, vec4 (used for cb_w)
__global__ void kCvtBf16(const float* __restrict__ src, unsigned short* __restrict__ dst) {
  size_t i = (size_t)blockIdx.x * blockDim.x + threadIdx.x;   // over N/4 vec4s
  f32x4 f = ((const f32x4*)src)[i];
  u32x2 r;
  r[0] = (unsigned)f2bf(f[0]) | ((unsigned)f2bf(f[1]) << 16);
  r[1] = (unsigned)f2bf(f[2]) | ((unsigned)f2bf(f[3]) << 16);
  ((u32x2*)dst)[i] = r;
}

// row norms of enc (z transposed): one thread per (b,t); d-loop coalesced.
__global__ void kEncNorm(const float* __restrict__ z, float* __restrict__ norm) {
  int n = blockIdx.x * blockDim.x + threadIdx.x;   // 0..B*T-1
  int b = n >> 9, t = n & (TT - 1);
  const float* p = z + (size_t)b * DIM * TT + t;
  float s = 0.f;
  for (int d = 0; d < DIM; ++d) { float x = p[(size_t)d * TT]; s += x * x; }
  norm[n] = fmaxf(sqrtf(s), EPSN);
}

// LDS-tiled transpose [B,D,T] -> enc_n bf16 [B*T, D] with L2 normalization
__global__ void kEncTrans(const float* __restrict__ z, const float* __restrict__ norm,
                          unsigned short* __restrict__ encn) {
  __shared__ float tile[32][33];
  int b = blockIdx.z, d0 = blockIdx.y * 32, t0 = blockIdx.x * 32;
  int tx = threadIdx.x, ty = threadIdx.y;           // 32 x 8
#pragma unroll
  for (int i = 0; i < 4; ++i) {
    int d = d0 + ty + 8 * i;
    tile[ty + 8 * i][tx] = z[((size_t)b * DIM + d) * TT + t0 + tx];
  }
  __syncthreads();
#pragma unroll
  for (int i = 0; i < 4; ++i) {
    int r = ty + 8 * i;                  // local t
    int n = b * TT + t0 + r;
    float inv = 1.0f / norm[n];
    encn[(size_t)n * DIM + d0 + tx] = f2bf(tile[tx][r] * inv);
  }
}

// ---------- K1: proj_cb = codebook @ cb_w^T + b (bf16 WMMA, fp32 acc) ----------
// wave tile: 16 (V) x 64 (D); double-buffered K loop so the next k-step's
// load clause overlaps the current 4 WMMAs instead of s_wait_loadcnt 0 stalls.
__global__ void kProjGemm(const float* __restrict__ codebook,
                          const unsigned short* __restrict__ cwbf,
                          const float* __restrict__ cb_b,
                          float* __restrict__ proj) {
  int wave = (int)((blockIdx.x * blockDim.x + threadIdx.x) >> 5);
  int lane = threadIdx.x & 31;
  int v0 = (wave >> 4) * 16;           // 2000 v-tiles
  int d0 = (wave & 15) * 64;           // 16 d-groups of 64
  v8f acc[4];
#pragma unroll
  for (int j = 0; j < 4; ++j) acc[j] = v8f{0.f,0.f,0.f,0.f,0.f,0.f,0.f,0.f};

  v16bf a_cur = ldfrag_f32(codebook, v0, CDI, 0, lane);
  v16bf b_cur[4];
#pragma unroll
  for (int j = 0; j < 4; ++j) b_cur[j] = ldfrag_bf16(cwbf, d0 + j * 16, CDI, 0, lane);

  for (int k0 = 0; k0 < CDI - 32; k0 += 32) {
    int kn = k0 + 32;
    __builtin_prefetch(codebook + (size_t)(v0 + (lane & 15)) * CDI + kn + 64, 0, 1);
    // issue next-step loads first (fills the clause, overlaps WMMA pipe)
    v16bf a_nxt = ldfrag_f32(codebook, v0, CDI, kn, lane);
    v16bf b_nxt[4];
#pragma unroll
    for (int j = 0; j < 4; ++j) b_nxt[j] = ldfrag_bf16(cwbf, d0 + j * 16, CDI, kn, lane);
#pragma unroll
    for (int j = 0; j < 4; ++j) acc[j] = WMMA_BF16(a_cur, b_cur[j], acc[j]);
    a_cur = a_nxt;
#pragma unroll
    for (int j = 0; j < 4; ++j) b_cur[j] = b_nxt[j];
  }
#pragma unroll
  for (int j = 0; j < 4; ++j) acc[j] = WMMA_BF16(a_cur, b_cur[j], acc[j]);

  // C layout: VGPR r, lane l -> (M = r + 8*(l>>4), N = l&15)
  int hi = (lane >> 4) & 1, nl = lane & 15;
#pragma unroll
  for (int j = 0; j < 4; ++j) {
    int d = d0 + j * 16 + nl;
    float bias = cb_b[d];
#pragma unroll
    for (int r = 0; r < 8; ++r) {
      int v = v0 + r + 8 * hi;
      proj[(size_t)v * DIM + d] = acc[j][r] + bias;
    }
  }
}

// Fused: row norm of proj_cb + normalized bf16 write (single pass over proj;
// row data stays in registers -> saves a full 131MB HBM read vs two kernels).
__global__ void kRowNormCvt(const float* __restrict__ proj,
                            unsigned short* __restrict__ cbn,
                            float* __restrict__ vnorm) {
  __shared__ float red[256];
  int v = blockIdx.x;
  f32x4 x = ((const f32x4*)(proj + (size_t)v * DIM))[threadIdx.x];  // 4 floats/thread
  float s = x[0]*x[0] + x[1]*x[1] + x[2]*x[2] + x[3]*x[3];
  red[threadIdx.x] = s; __syncthreads();
  for (int st = 128; st > 0; st >>= 1) {
    if (threadIdx.x < st) red[threadIdx.x] += red[threadIdx.x + st];
    __syncthreads();
  }
  float nrm = fmaxf(sqrtf(red[0]), EPSN);
  if (threadIdx.x == 0) vnorm[v] = nrm;
  float inv = 1.0f / nrm;
  u32x2 r;
  r[0] = (unsigned)f2bf(x[0] * inv) | ((unsigned)f2bf(x[1] * inv) << 16);
  r[1] = (unsigned)f2bf(x[2] * inv) | ((unsigned)f2bf(x[3] * inv) << 16);
  ((u32x2*)(cbn + (size_t)v * DIM))[threadIdx.x] = r;
}

__global__ void kInitBest(unsigned long long* __restrict__ best) {
  int n = blockIdx.x * blockDim.x + threadIdx.x;
  best[n] = 0ull;   // below any ford() key
}

// ---------- K2: fused similarity GEMM + argmax ----------
// wave: fixed 16-row n-tile, scans 16 v-tiles (groups of 4 accumulators),
// double-buffered K loop, register-resident running best, lane-pair combine,
// one packed u64 atomicMax per n.
__global__ void kSimArgmax(const unsigned short* __restrict__ cbn,
                           const unsigned short* __restrict__ encn,
                           unsigned long long* __restrict__ best) {
  int wave = (int)((blockIdx.x * blockDim.x + threadIdx.x) >> 5);
  int lane = threadIdx.x & 31;
  int ntile = wave / 125;              // 128 n-tiles
  int chunk = wave % 125;              // 125 chunks x 16 v-tiles = 2000
  int n0 = ntile * 16;
  int hi = (lane >> 4) & 1, nl = lane & 15;
  float bval = -3.0e38f; unsigned int bidx = 0u;

  for (int g = 0; g < 4; ++g) {
    int vt0 = chunk * 16 + g * 4;
    v8f acc[4];
#pragma unroll
    for (int j = 0; j < 4; ++j) acc[j] = v8f{0.f,0.f,0.f,0.f,0.f,0.f,0.f,0.f};

    v16bf b_cur = ldfrag_bf16(encn, n0, DIM, 0, lane);   // N side = enc rows
    v16bf a_cur[4];
#pragma unroll
    for (int j = 0; j < 4; ++j) a_cur[j] = ldfrag_bf16(cbn, (vt0 + j) * 16, DIM, 0, lane);

    for (int k0 = 0; k0 < DIM - 32; k0 += 32) {
      int kn = k0 + 32;
      __builtin_prefetch(cbn + (size_t)(vt0 * 16 + (lane & 15)) * DIM + kn + 64, 0, 1);
      v16bf b_nxt = ldfrag_bf16(encn, n0, DIM, kn, lane);
      v16bf a_nxt[4];
#pragma unroll
      for (int j = 0; j < 4; ++j) a_nxt[j] = ldfrag_bf16(cbn, (vt0 + j) * 16, DIM, kn, lane);
#pragma unroll
      for (int j = 0; j < 4; ++j) acc[j] = WMMA_BF16(a_cur[j], b_cur, acc[j]);
      b_cur = b_nxt;
#pragma unroll
      for (int j = 0; j < 4; ++j) a_cur[j] = a_nxt[j];
    }
#pragma unroll
    for (int j = 0; j < 4; ++j) acc[j] = WMMA_BF16(a_cur[j], b_cur, acc[j]);

#pragma unroll
    for (int j = 0; j < 4; ++j)
#pragma unroll
      for (int r = 0; r < 8; ++r) {
        float s = acc[j][r];
        unsigned int v = (unsigned)((vt0 + j) * 16 + r + 8 * hi);
        if (s > bval || (s == bval && v < bidx)) { bval = s; bidx = v; }
      }
  }
  // lanes l and l^16 hold the same n (different M halves): combine
  float oval = __shfl_xor(bval, 16, 32);
  unsigned int oidx = (unsigned int)__shfl_xor((int)bidx, 16, 32);
  if (oval > bval || (oval == bval && oidx < bidx)) { bval = oval; bidx = oidx; }
  // packed key: ordered float high 32, inverted index low (ties -> lowest idx)
  unsigned long long key =
      ((unsigned long long)ford(bval) << 32) | (unsigned long long)(0xFFFFFFFFu - bidx);
  if (lane < 16) atomicMax(&best[n0 + nl], key);
}

__global__ void kDecodeIdx(const unsigned long long* __restrict__ best,
                           int* __restrict__ idx, float* __restrict__ out_idx,
                           float* __restrict__ out_loss) {
  int n = blockIdx.x * blockDim.x + threadIdx.x;   // 2048
  unsigned int v = 0xFFFFFFFFu - (unsigned int)(best[n] & 0xFFFFFFFFull);
  idx[n] = (int)v;
  out_idx[n] = (float)v;
  if (n < 8) out_loss[n] = 0.f;                    // zero both loss vectors
}

// ---------- K3: gather z_q, write straight-through output, losses ----------
__global__ void kGatherLoss(const float* __restrict__ proj, const int* __restrict__ idx,
                            const float* __restrict__ z, float* __restrict__ out_zq,
                            float* __restrict__ out_loss) {
  __shared__ float red[256];
  size_t id = (size_t)blockIdx.x * 256 + threadIdx.x;   // B*D*T
  int t = (int)(id & (TT - 1));
  int d = (int)((id >> 9) & (DIM - 1));
  int b = (int)(id >> 19);
  int code = idx[b * TT + t];
  float q = proj[(size_t)code * DIM + d];
  out_zq[id] = q;                                        // z_q_st == z_q numerically
  float diff = z[id] - q;
  red[threadIdx.x] = diff * diff;
  __syncthreads();
  for (int s = 128; s > 0; s >>= 1) {
    if (threadIdx.x < s) red[threadIdx.x] += red[threadIdx.x + s];
    __syncthreads();
  }
  if (threadIdx.x == 0) {
    float partial = red[0] * (1.0f / ((float)DIM * (float)TT));
    atomicAdd(&out_loss[b], partial);        // commitment_loss[b]
    atomicAdd(&out_loss[4 + b], partial);    // codebook_loss[b] (numerically equal)
  }
}

extern "C" void kernel_launch(void* const* d_in, const int* in_sizes, int n_in,
                              void* d_out, int out_size, void* d_ws, size_t ws_size,
                              hipStream_t stream) {
  const float* z        = (const float*)d_in[0];   // [B,D,T]
  const float* codebook = (const float*)d_in[1];   // [V,CD]
  const float* cb_w     = (const float*)d_in[2];   // [D,CD]
  const float* cb_b     = (const float*)d_in[3];   // [D]

  float* out      = (float*)d_out;
  float* out_zq   = out;                                    // B*D*T
  float* out_idx  = out + (size_t)BB * DIM * TT;            // B*T
  float* out_loss = out_idx + (size_t)BB * TT;              // 8

  // Workspace carve-up (~209.4 MB total)
  char* ws = (char*)d_ws;
  unsigned short* cwbf = (unsigned short*)(ws);                        // D*CD bf16   = 8,388,608 B
  unsigned short* encn = (unsigned short*)(ws + 8388608);              // B*T*D bf16  = 4,194,304 B
  float*          proj = (float*)        (ws + 12582912);              // V*D fp32    = 131,072,000 B
  unsigned short* cbn  = (unsigned short*)(ws + 143654912);            // V*D bf16    = 65,536,000 B
  float*          vnorm= (float*)        (ws + 209190912);             // V fp32
  float*          enorm= (float*)        (ws + 209318912);             // B*T fp32
  unsigned long long* best = (unsigned long long*)(ws + 209327104);    // B*T u64
  int*            idx  = (int*)          (ws + 209343488);             // B*T i32

  // 1) cb_w -> bf16 (reused 2000x in K1)
  kCvtBf16<<<dim3((DIM * CDI / 4) / 256), dim3(256), 0, stream>>>(cb_w, cwbf);
  // 2) enc row norms, 3) normalize+transpose -> enc_n bf16
  kEncNorm<<<dim3((BB * TT) / 256), dim3(256), 0, stream>>>(z, enorm);
  kEncTrans<<<dim3(TT / 32, DIM / 32, BB), dim3(32, 8), 0, stream>>>(z, enorm, encn);
  // 4) projection GEMM: 2000 v-tiles x 16 d-groups = 32000 waves = 4000 blocks
  kProjGemm<<<dim3(4000), dim3(256), 0, stream>>>(codebook, cwbf, cb_b, proj);
  // 5+6) fused proj row norms + normalized bf16 codebook (single pass)
  kRowNormCvt<<<dim3(VV), dim3(256), 0, stream>>>(proj, cbn, vnorm);
  // 7) reset per-row best keys
  kInitBest<<<dim3((BB * TT) / 256), dim3(256), 0, stream>>>(best);
  // 8) fused similarity GEMM + argmax: 128 n-tiles x 125 chunks = 16000 waves
  kSimArgmax<<<dim3(2000), dim3(256), 0, stream>>>(cbn, encn, best);
  // 9) decode indices (+ zero loss accumulators), 10) gather + losses
  kDecodeIdx<<<dim3((BB * TT) / 256), dim3(256), 0, stream>>>(best, idx, out_idx, out_loss);
  kGatherLoss<<<dim3((BB * DIM * TT) / 256), dim3(256), 0, stream>>>(proj, idx, z, out_zq, out_loss);
}
